// RGPNet_27608049779105
// MI455X (gfx1250) — matrix operand order repaired and checked by reading, not compiled, not run on hardware
//
#include <hip/hip_runtime.h>
#include <math.h>

// ---------------------------------------------------------------------------
// RGPNet-like net for MI455X (gfx1250, wave32, WMMA).
// All heavy math via v_wmma_f32_16x16x32_f16 (f16 in, f32 acc).
// Working set (~76MB f16 activations + 16MB f16 weights) is L2-resident
// (192MB), so kernels are matrix-throughput bound, not HBM bound.
//
// Branch-free fragment feeds:
//  - A (weights):  two 16B vector loads from K-padded packed rows.
//  - GEMM B:       one 32B load from fragment-ordered packed FC weights.
//  - conv B:       activations live in halo-padded (C,H+2p,W+2p) tensors with
//                  zeroed borders; im2col gather = ds_load_b128 offsets +
//                  v_add(posOff) + global_load_u16, no range checks.
//  - conv waves register-block CTB Cout-tiles per spatial tile, so each
//                  B gather feeds CTB WMMAs.
// Workspace requirement: ~172 MB.
// ---------------------------------------------------------------------------

typedef __attribute__((ext_vector_type(16))) _Float16 v16h;
typedef __attribute__((ext_vector_type(8)))  _Float16 v8h;
typedef __attribute__((ext_vector_type(8)))  float    v8f;

#define DEVINL __device__ __forceinline__

DEVINL v8f wmma_f16(v16h a, v16h b, v8f c) {
  // (neg_a, A, neg_b, B, c_mod, C, reuse_a, reuse_b)
  return __builtin_amdgcn_wmma_f32_16x16x32_f16(false, a, false, b, (short)0, c,
                                                false, false);
}

DEVINL v16h cat8(v8h lo, v8h hi) {
  return __builtin_shufflevector(lo, hi, 0, 1, 2, 3, 4, 5, 6, 7,
                                 8, 9, 10, 11, 12, 13, 14, 15);
}

DEVINL float lrelu(float x, float al) { return x >= 0.f ? x : al * x; }

// ---------------------------------------------------------------------------
// silho staging: f32 (256,1,64,44) -> f16 halo-2 padded (256,1,68,48)
// ---------------------------------------------------------------------------
__global__ void k_stage_sil(const float* __restrict__ s, _Float16* __restrict__ d) {
  const int total = 256 * 68 * 48;
  int i = blockIdx.x * blockDim.x + threadIdx.x;
  int stride = gridDim.x * blockDim.x;
  for (; i < total; i += stride) {
    int x = i % 48;
    int r = i / 48;
    int y = r % 68;
    int img = r / 68;
    _Float16 v = (_Float16)0.f;
    if (y >= 2 && y < 66 && x >= 2 && x < 46)
      v = (_Float16)s[img * 2816 + (y - 2) * 44 + (x - 2)];
    d[i] = v;
  }
}

// ---------------------------------------------------------------------------
// Conv weight pack: f32 (Cout,K) -> f16 (Cout,Kpad) zero-padded rows
// (Kpad % 32 == 0, rows 64B-aligned -> A fragments are two b128 loads).
// ---------------------------------------------------------------------------
__global__ void k_pack_wconv(const float* __restrict__ src, _Float16* __restrict__ dst,
                             int Cout, int K, int Kpad) {
  int total = Cout * Kpad;
  int i = blockIdx.x * blockDim.x + threadIdx.x;
  int stride = gridDim.x * blockDim.x;
  for (; i < total; i += stride) {
    int c = i / Kpad, k = i - c * Kpad;
    dst[i] = (k < K) ? (_Float16)src[c * K + k] : (_Float16)0.f;
  }
}

// ---------------------------------------------------------------------------
// FC weight pack: f32 (K,N) row-major -> f16 fragment order so each lane's
// B fragment is one contiguous 32B load.
// ---------------------------------------------------------------------------
__global__ void k_pack_bfc(const float* __restrict__ src, _Float16* __restrict__ dst,
                           int N, int K) {
  int nkc = K >> 5;
  int total = (N >> 4) * nkc * 512;
  int idx = blockIdx.x * blockDim.x + threadIdx.x;
  int stride = gridDim.x * blockDim.x;
  for (; idx < total; idx += stride) {
    int i    = idx & 15;
    int lane = (idx >> 4) & 31;
    int rest = idx >> 9;
    int kc   = rest % nkc;
    int ct   = rest / nkc;
    int k    = (kc << 5) + ((lane >= 16) ? 16 : 0) + i;
    int col  = (ct << 4) + (lane & 15);
    dst[idx] = (_Float16)src[(size_t)k * N + col];
  }
}

// ---------------------------------------------------------------------------
// Conv (stride 1, KSxKS, zero pad = KS/2) + leaky-ReLU, implicit-GEMM WMMA.
// Input:  halo-padded (Cin, H+2p, W+2p) with zero borders.
// Output: (Cout, H+2*outHalo, W+2*outHalo); this kernel zeroes its own halo.
// One block (8 waves) per image; each wave owns CTB 16(Cout)x16(spatial)
// tiles sharing one B gather per k-chunk.
// ---------------------------------------------------------------------------
template <int KS, int CTB>
__global__ void __launch_bounds__(256)
k_conv_lrelu_wmma(const _Float16* __restrict__ in, const _Float16* __restrict__ wp,
                  _Float16* __restrict__ out,
                  int Cin, int Cout, int H, int W, int Kpad, int outHalo) {
  constexpr int KK  = KS * KS;
  constexpr int pad = KS / 2;
  __shared__ __align__(16) int offTab[1184];

  const int img    = blockIdx.x;
  const int lane   = threadIdx.x & 31;
  const int wave   = threadIdx.x >> 5;
  const int nwaves = blockDim.x >> 5;
  const bool hi    = lane >= 16;
  const int  off   = hi ? 8 : 0;
  const int  hioff = hi ? 16 : 0;
  const int  Hip   = H + 2 * pad, Wip = W + 2 * pad;
  const int  HWip  = Hip * Wip;
  const int  Hop   = H + 2 * outHalo, Wop = W + 2 * outHalo;
  const int  HWop  = Hop * Wop;
  const int  HW    = H * W;
  const int  K     = Cin * KK;
  const int  spT   = HW >> 4;
  const int  nct   = (Cout >> 4) / CTB;
  const int  total = spT * nct;
  const _Float16* inImg = in + (size_t)img * Cin * HWip;
  _Float16* outImg      = out + (size_t)img * Cout * HWop;

  // k -> flat padded-input offset (tail k >= K maps to 0; A weights are zero there)
  for (int k = threadIdx.x; k < Kpad; k += blockDim.x) {
    int v = 0;
    if (k < K) {
      int ci = k / KK;
      int r  = k - ci * KK;
      int ky = r / KS;
      int kx = r - ky * KS;
      v = ci * HWip + ky * Wip + kx;
    }
    offTab[k] = v;
  }
  // zero this image's output halo (disjoint from interior writes; no sync vs them)
  if (outHalo) {
    int bpc = 2 * Wop + 2 * (Hop - 2);
    for (int idx = threadIdx.x; idx < Cout * bpc; idx += blockDim.x) {
      int c = idx / bpc, e = idx - c * bpc;
      int y, x;
      if (e < Wop)            { y = 0;       x = e; }
      else if (e < 2 * Wop)   { y = Hop - 1; x = e - Wop; }
      else { int e2 = e - 2 * Wop; y = 1 + (e2 >> 1); x = (e2 & 1) ? (Wop - 1) : 0; }
      outImg[(size_t)c * HWop + y * Wop + x] = (_Float16)0.f;
    }
  }
  __syncthreads();

  for (int t = wave; t < total; t += nwaves) {
    const int cg    = t / spT;
    const int st    = t - cg * spT;
    const int cout0 = cg * (CTB * 16);
    const int p     = (st << 4) + (lane & 15);
    const int py    = p / W;
    const int pxl   = p - py * W;
    const _Float16* bp = inImg + py * Wip + pxl;       // loop-invariant per tile
    const _Float16* wrow0 = wp + (size_t)(cout0 + (lane & 15)) * Kpad;
    v8f acc[CTB];
    v8f z = {};
#pragma unroll
    for (int c = 0; c < CTB; ++c) acc[c] = z;
    for (int k0 = 0; k0 < Kpad; k0 += 32) {
      __builtin_prefetch(wrow0 + k0 + 32, 0, 1);       // global_prefetch_b8
      // B gather once per chunk: 4x ds_load_b128 offsets + 16 u16 loads
      const int* ot = offTab + k0 + hioff;
      int4 o0 = *(const int4*)(ot + 0);
      int4 o1 = *(const int4*)(ot + 4);
      int4 o2 = *(const int4*)(ot + 8);
      int4 o3 = *(const int4*)(ot + 12);
      v16h b;
      b[0]  = bp[o0.x]; b[1]  = bp[o0.y]; b[2]  = bp[o0.z]; b[3]  = bp[o0.w];
      b[4]  = bp[o1.x]; b[5]  = bp[o1.y]; b[6]  = bp[o1.z]; b[7]  = bp[o1.w];
      b[8]  = bp[o2.x]; b[9]  = bp[o2.y]; b[10] = bp[o2.z]; b[11] = bp[o2.w];
      b[12] = bp[o3.x]; b[13] = bp[o3.y]; b[14] = bp[o3.z]; b[15] = bp[o3.w];
#pragma unroll
      for (int c = 0; c < CTB; ++c) {
        const _Float16* wrow = wrow0 + (size_t)c * 16 * Kpad;
        v8h alo = *(const v8h*)(wrow + k0 + hioff);
        v8h ahi = *(const v8h*)(wrow + k0 + hioff + 8);
        acc[c] = wmma_f16(cat8(alo, ahi), b, acc[c]);
      }
    }
#pragma unroll
    for (int c = 0; c < CTB; ++c) {
#pragma unroll
      for (int i = 0; i < 8; ++i) {
        int m = i + (hi ? 8 : 0);
        outImg[(size_t)(cout0 + c * 16 + m) * HWop +
               (py + outHalo) * Wop + (pxl + outHalo)] =
            (_Float16)lrelu(acc[c][i], 0.01f);
      }
    }
  }
}

// ---------------------------------------------------------------------------
// GEMM + bias: C[M,N] f32 = A[M,K] f16 @ Bpacked + bias.  One wave per 16x16
// D tile; A rows clamped (rows >= M never stored); inner loop = 3 vector
// loads + 1 WMMA, unrolled x4.
// ---------------------------------------------------------------------------
__global__ void __launch_bounds__(32)
k_gemm_bias_wmma(const _Float16* __restrict__ A, const _Float16* __restrict__ Bp,
                 const float* __restrict__ bias, float* __restrict__ C,
                 int M, int N, int K) {
  const int lane = threadIdx.x & 31;
  const bool hi  = lane >= 16;
  const int off  = hi ? 8 : 0;
  const int row0 = blockIdx.x << 4;
  int arow = row0 + (lane & 15);
  if (arow > M - 1) arow = M - 1;
  const _Float16* Arow = A + (size_t)arow * K;
  const int nkc = K >> 5;
  const _Float16* Bt = Bp + (size_t)blockIdx.y * nkc * 512 + (size_t)lane * 16;
  v8f acc = {};
#pragma unroll 4
  for (int kc = 0; kc < nkc; ++kc) {
    v8h alo = *(const v8h*)(Arow + (kc << 5) + 2 * off);
    v8h ahi = *(const v8h*)(Arow + (kc << 5) + 2 * off + 8);
    v16h a = cat8(alo, ahi);
    v16h b = *(const v16h*)(Bt + (size_t)kc * 512);
    acc = wmma_f16(a, b, acc);
  }
  const int col = (blockIdx.y << 4) + (lane & 15);
#pragma unroll
  for (int i = 0; i < 8; ++i) {
    int r = row0 + i + (hi ? 8 : 0);
    if (r < M) C[(size_t)r * N + col] = acc[i] + bias[col];
  }
}

// ---------------------------------------------------------------------------
// 2x2 max-pool (valid, stride 2).  Input (NC,H+2,W+2) halo1, output
// (NC,H/2+2,W/2+2) halo1 with zero border written by this kernel.
// ---------------------------------------------------------------------------
__global__ void k_pool2(const _Float16* __restrict__ in, _Float16* __restrict__ out,
                        int NC, int H, int W) {
  const int OH = H >> 1, OW = W >> 1;
  const int Wip = W + 2;
  const int Hop = OH + 2, Wop = OW + 2;
  long total = (long)NC * Hop * Wop;
  long i = (long)blockIdx.x * blockDim.x + threadIdx.x;
  long stride = (long)gridDim.x * blockDim.x;
  for (; i < total; i += stride) {
    int x = (int)(i % Wop);
    long r = i / Wop;
    int y = (int)(r % Hop);
    int nc = (int)(r / Hop);
    _Float16 v = (_Float16)0.f;
    if (y > 0 && y < Hop - 1 && x > 0 && x < Wop - 1) {
      int oy = y - 1, ox = x - 1;
      const _Float16* p = in + ((size_t)nc * (H + 2) + 2 * oy + 1) * Wip + 2 * ox + 1;
      float a = (float)p[0], b = (float)p[1], c = (float)p[Wip], d = (float)p[Wip + 1];
      v = (_Float16)fmaxf(fmaxf(a, b), fmaxf(c, d));
    }
    out[i] = v;
  }
}

// ---------------------------------------------------------------------------
// max over sequence dim: (B,S,F) f16 -> (B,F) f16
// ---------------------------------------------------------------------------
__global__ void k_max_over_s(const _Float16* __restrict__ a, _Float16* __restrict__ gx,
                             int Bn, int Sn, int F) {
  int i = blockIdx.x * blockDim.x + threadIdx.x;
  int total = Bn * F;
  int stride = gridDim.x * blockDim.x;
  for (; i < total; i += stride) {
    int b = i / F, k = i - b * F;
    float m = -1e30f;
    for (int s = 0; s < Sn; ++s)
      m = fmaxf(m, (float)a[((size_t)(b * Sn + s)) * F + k]);
    gx[i] = (_Float16)m;
  }
}

// ---------------------------------------------------------------------------
// ROI crop (14x14 at clamped center-7) + bilinear upsample to 32x32 (half-
// pixel centers, matches jax.image.resize 'linear' upsampling).  Output:
// halo-2 padded f16 images (1024,1,36,36), img = (b*4 + r)*32 + s.
// ---------------------------------------------------------------------------
__global__ void k_roi(const float* __restrict__ silho, const float* __restrict__ px,
                      _Float16* __restrict__ out) {
  const int total = 1024 * 36 * 36;
  int i = blockIdx.x * blockDim.x + threadIdx.x;
  int stride = gridDim.x * blockDim.x;
  for (; i < total; i += stride) {
    int oxp = i % 36;
    int r1  = i / 36;
    int oyp = r1 % 36;
    int img = r1 / 36;
    _Float16 result = (_Float16)0.f;
    if (oyp >= 2 && oyp < 34 && oxp >= 2 && oxp < 34) {
      int oy = oyp - 2, ox = oxp - 2;
      int s  = img & 31;
      int br = img >> 5;
      int rr = br & 3;
      int b  = br >> 2;
      float c0 = px[(b * 4 + rr) * 2 + 0];
      float c1 = px[(b * 4 + rr) * 2 + 1];
      int st0 = (int)c0 - 7; st0 = st0 < 0 ? 0 : (st0 > 50 ? 50 : st0);
      int st1 = (int)c1 - 7; st1 = st1 < 0 ? 0 : (st1 > 30 ? 30 : st1);
      const float* im = silho + ((size_t)b * 32 + s) * (64 * 44);
      float fy = (oy + 0.5f) * (14.f / 32.f) - 0.5f;
      float fx = (ox + 0.5f) * (14.f / 32.f) - 0.5f;
      int y0 = (int)floorf(fy); float wy = fy - (float)y0;
      int x0 = (int)floorf(fx); float wx = fx - (float)x0;
      int ya = y0     < 0 ? 0 : (y0     > 13 ? 13 : y0);
      int yb = y0 + 1 < 0 ? 0 : (y0 + 1 > 13 ? 13 : y0 + 1);
      int xa = x0     < 0 ? 0 : (x0     > 13 ? 13 : x0);
      int xb = x0 + 1 < 0 ? 0 : (x0 + 1 > 13 ? 13 : x0 + 1);
      float v00 = im[(st0 + ya) * 44 + st1 + xa];
      float v01 = im[(st0 + ya) * 44 + st1 + xb];
      float v10 = im[(st0 + yb) * 44 + st1 + xa];
      float v11 = im[(st0 + yb) * 44 + st1 + xb];
      result = (_Float16)(v00 * (1.f - wy) * (1.f - wx) + v01 * (1.f - wy) * wx +
                          v10 * wy * (1.f - wx)         + v11 * wy * wx);
    }
    out[i] = result;
  }
}

// ---------------------------------------------------------------------------
// 2-layer GAT on 128 nodes (tiny: VALU + LDS). One block per batch element,
// one thread per node.  Masked softmax is branch-free (cndmask selects).
// Output: elu -> log_softmax over 16 features.
// ---------------------------------------------------------------------------
DEVINL bool adj_fn(int i, int j) {
  int di = (i >> 2) - (j >> 2);
  bool same    = (di == 0);
  bool corners = ((i & 3) | (j & 3)) == 0;
  bool nb      = corners & ((di == 1) | (di == -1));
  return same | nb;
}

__global__ void __launch_bounds__(128)
k_gat(const float* __restrict__ lfc,
      const float* __restrict__ W1, const float* __restrict__ a11, const float* __restrict__ a12,
      const float* __restrict__ W2, const float* __restrict__ a21, const float* __restrict__ a22,
      float* __restrict__ gat_out) {
  __shared__ float hbuf[128 * 8];
  __shared__ float h2buf[128 * 16];
  __shared__ float sb1[128];
  __shared__ float sb2[128];
  const int b = blockIdx.x;
  const int i = threadIdx.x;
  const float* xb = lfc + (size_t)b * 128 * 256;

  float h[8];
#pragma unroll
  for (int f = 0; f < 8; ++f) h[f] = 0.f;
  for (int k = 0; k < 256; ++k) {
    float xv = xb[i * 256 + k];
#pragma unroll
    for (int f = 0; f < 8; ++f) h[f] += xv * W1[k * 8 + f];
  }
  float s1 = 0.f, s2 = 0.f;
#pragma unroll
  for (int f = 0; f < 8; ++f) { hbuf[i * 8 + f] = h[f]; s1 += h[f] * a11[f]; s2 += h[f] * a12[f]; }
  sb1[i] = s1; sb2[i] = s2;
  __syncthreads();

  float h1[8];
  {
    float si = sb1[i];
    float mx = -1e30f;
    for (int j = 0; j < 128; ++j) {
      float e = lrelu(si + sb2[j], 0.2f);
      mx = fmaxf(mx, adj_fn(i, j) ? e : -1e30f);
    }
    float den = 0.f;
    float acc[8];
#pragma unroll
    for (int f = 0; f < 8; ++f) acc[f] = 0.f;
    for (int j = 0; j < 128; ++j) {
      float e = lrelu(si + sb2[j], 0.2f);
      float w = adj_fn(i, j) ? expf(e - mx) : 0.f;
      den += w;
#pragma unroll
      for (int f = 0; f < 8; ++f) acc[f] += w * hbuf[j * 8 + f];
    }
    float inv = 1.f / den;
#pragma unroll
    for (int f = 0; f < 8; ++f) h1[f] = acc[f] * inv;
  }
  __syncthreads();

  float g[16];
#pragma unroll
  for (int f2 = 0; f2 < 16; ++f2) {
    float v = 0.f;
#pragma unroll
    for (int f = 0; f < 8; ++f) v += h1[f] * W2[f * 16 + f2];
    g[f2] = v;
    h2buf[i * 16 + f2] = v;
  }
  s1 = 0.f; s2 = 0.f;
#pragma unroll
  for (int f2 = 0; f2 < 16; ++f2) { s1 += g[f2] * a21[f2]; s2 += g[f2] * a22[f2]; }
  sb1[i] = s1; sb2[i] = s2;
  __syncthreads();

  float o[16];
  {
    float si = sb1[i];
    float mx = -1e30f;
    for (int j = 0; j < 128; ++j) {
      float e = lrelu(si + sb2[j], 0.2f);
      mx = fmaxf(mx, adj_fn(i, j) ? e : -1e30f);
    }
    float den = 0.f;
    float acc[16];
#pragma unroll
    for (int f2 = 0; f2 < 16; ++f2) acc[f2] = 0.f;
    for (int j = 0; j < 128; ++j) {
      float e = lrelu(si + sb2[j], 0.2f);
      float w = adj_fn(i, j) ? expf(e - mx) : 0.f;
      den += w;
#pragma unroll
      for (int f2 = 0; f2 < 16; ++f2) acc[f2] += w * h2buf[j * 16 + f2];
    }
    float inv = 1.f / den;
#pragma unroll
    for (int f2 = 0; f2 < 16; ++f2) {
      float v = acc[f2] * inv;
      o[f2] = v > 0.f ? v : (expf(v) - 1.f);   // elu
    }
  }
  float mx = -1e30f;
#pragma unroll
  for (int f2 = 0; f2 < 16; ++f2) mx = fmaxf(mx, o[f2]);
  float lse = 0.f;
#pragma unroll
  for (int f2 = 0; f2 < 16; ++f2) lse += expf(o[f2] - mx);
  lse = logf(lse) + mx;
#pragma unroll
  for (int f2 = 0; f2 < 16; ++f2)
    gat_out[((size_t)b * 128 + i) * 16 + f2] = o[f2] - lse;
}

// ---------------------------------------------------------------------------
// feature = concat([g_fc, max_i l_fc, max_i gat]) -> (8,1,528) f32
// ---------------------------------------------------------------------------
__global__ void k_finalize(const float* __restrict__ gfc, const float* __restrict__ lfc,
                           const float* __restrict__ gat, float* __restrict__ out) {
  int b = blockIdx.x;
  int j = threadIdx.x;
  if (j >= 528) return;
  float v;
  if (j < 256) {
    v = gfc[b * 256 + j];
  } else if (j < 512) {
    int c = j - 256;
    v = -1e30f;
    for (int i = 0; i < 128; ++i)
      v = fmaxf(v, lfc[((size_t)b * 128 + i) * 256 + c]);
  } else {
    int c = j - 512;
    v = -1e30f;
    for (int i = 0; i < 128; ++i)
      v = fmaxf(v, gat[((size_t)b * 128 + i) * 16 + c]);
  }
  out[b * 528 + j] = v;
}

// ---------------------------------------------------------------------------
// host orchestration
// ---------------------------------------------------------------------------
extern "C" void kernel_launch(void* const* d_in, const int* in_sizes, int n_in,
                              void* d_out, int out_size, void* d_ws, size_t ws_size,
                              hipStream_t stream) {
  (void)in_sizes; (void)n_in; (void)out_size; (void)ws_size;
  const float* silho = (const float*)d_in[0];
  const float* px    = (const float*)d_in[1];
  const float* w[6]  = {(const float*)d_in[2], (const float*)d_in[3], (const float*)d_in[4],
                        (const float*)d_in[5], (const float*)d_in[6], (const float*)d_in[7]};
  const float* fcgw = (const float*)d_in[8];
  const float* fcgb = (const float*)d_in[9];
  const float* fclw = (const float*)d_in[10];
  const float* fclb = (const float*)d_in[11];
  const float* g1W  = (const float*)d_in[12];
  const float* g1a1 = (const float*)d_in[13];
  const float* g1a2 = (const float*)d_in[14];
  const float* g2W  = (const float*)d_in[15];
  const float* g2a1 = (const float*)d_in[16];
  const float* g2a2 = (const float*)d_in[17];
  float* out = (float*)d_out;

  // -------- workspace layout (~172 MB) --------
  char* ws = (char*)d_ws;
  size_t off = 0;
  auto take = [&](size_t bytes) -> char* {
    char* p = ws + off;
    off = (off + bytes + 255) & ~(size_t)255;
    return p;
  };
  _Float16* sil16p = (_Float16*)take((size_t)256 * 68 * 48 * 2);   // halo-2 padded
  const int cK[6]  = {25, 288, 288, 576, 576, 1152};
  const int cKp[6] = {32, 288, 288, 576, 576, 1152};
  const int cCo[6] = {32, 32, 64, 64, 128, 128};
  _Float16* wp[6];
  for (int i = 0; i < 6; ++i) wp[i] = (_Float16*)take((size_t)cCo[i] * cKp[i] * 2);
  _Float16* fcgwp = (_Float16*)take((size_t)22528 * 256 * 2);      // fragment order
  _Float16* fclwp = (_Float16*)take((size_t)8192 * 256 * 2);       // fragment order
  _Float16* gx16  = (_Float16*)take((size_t)8 * 22528 * 2);
  _Float16* roi16p = (_Float16*)take((size_t)1024 * 36 * 36 * 2);  // halo-2 padded
  float*    gfc   = (float*)take((size_t)8 * 256 * 4);
  float*    lfc   = (float*)take((size_t)1024 * 256 * 4);
  float*    gat   = (float*)take((size_t)8 * 128 * 16 * 4);
  const size_t BUFSZ = (size_t)1024 * 32 * 34 * 34;                // 37.9M halves
  _Float16* bufA = (_Float16*)take(BUFSZ * 2);
  _Float16* bufB = (_Float16*)take(BUFSZ * 2);

  auto pool = [&](const _Float16* in, _Float16* o, int NC, int H, int W) {
    long tot = (long)NC * (H / 2 + 2) * (W / 2 + 2);
    int blocks = (int)((tot + 255) / 256);
    if (blocks > 8192) blocks = 8192;
    k_pool2<<<blocks, 256, 0, stream>>>(in, o, NC, H, W);
  };

  // staging + weight packing
  k_stage_sil<<<2048, 256, 0, stream>>>(silho, sil16p);
  for (int i = 0; i < 6; ++i) {
    int tot = cCo[i] * cKp[i];
    k_pack_wconv<<<(tot + 255) / 256, 256, 0, stream>>>(w[i], wp[i], cCo[i], cK[i], cKp[i]);
  }
  k_pack_bfc<<<4096, 256, 0, stream>>>(fcgw, fcgwp, 256, 22528);
  k_pack_bfc<<<4096, 256, 0, stream>>>(fclw, fclwp, 256, 8192);

  // -------- global branch: 256 images of 64x44 --------
  k_conv_lrelu_wmma<5, 2><<<256, 256, 0, stream>>>(sil16p, wp[0], bufA, 1,   32, 64, 44, cKp[0], 1);
  k_conv_lrelu_wmma<3, 2><<<256, 256, 0, stream>>>(bufA,   wp[1], bufB, 32,  32, 64, 44, cKp[1], 1);
  pool(bufB, bufA, 256 * 32, 64, 44);                               // -> 32x22 (halo1)
  k_conv_lrelu_wmma<3, 4><<<256, 256, 0, stream>>>(bufA,   wp[2], bufB, 32,  64, 32, 22, cKp[2], 1);
  k_conv_lrelu_wmma<3, 4><<<256, 256, 0, stream>>>(bufB,   wp[3], bufA, 64,  64, 32, 22, cKp[3], 1);
  pool(bufA, bufB, 256 * 64, 32, 22);                               // -> 16x11 (halo1)
  k_conv_lrelu_wmma<3, 4><<<256, 256, 0, stream>>>(bufB,   wp[4], bufA, 64, 128, 16, 11, cKp[4], 1);
  k_conv_lrelu_wmma<3, 4><<<256, 256, 0, stream>>>(bufA,   wp[5], bufB, 128,128, 16, 11, cKp[5], 0);
  k_max_over_s<<<(8 * 22528 + 255) / 256, 256, 0, stream>>>(bufB, gx16, 8, 32, 22528);
  k_gemm_bias_wmma<<<dim3(1, 16), 32, 0, stream>>>(gx16, fcgwp, fcgb, gfc, 8, 256, 22528);

  // -------- local branch: ROI -> 1024 images of 32x32 --------
  k_roi<<<2048, 256, 0, stream>>>(silho, px, roi16p);
  k_conv_lrelu_wmma<5, 2><<<1024, 256, 0, stream>>>(roi16p, wp[0], bufA, 1,   32, 32, 32, cKp[0], 1);
  k_conv_lrelu_wmma<3, 2><<<1024, 256, 0, stream>>>(bufA,   wp[1], bufB, 32,  32, 32, 32, cKp[1], 1);
  pool(bufB, bufA, 1024 * 32, 32, 32);                              // -> 16x16 (halo1)
  k_conv_lrelu_wmma<3, 4><<<1024, 256, 0, stream>>>(bufA,   wp[2], bufB, 32,  64, 16, 16, cKp[2], 1);
  k_conv_lrelu_wmma<3, 4><<<1024, 256, 0, stream>>>(bufB,   wp[3], bufA, 64,  64, 16, 16, cKp[3], 1);
  pool(bufA, bufB, 1024 * 64, 16, 16);                              // -> 8x8 (halo1)
  k_conv_lrelu_wmma<3, 4><<<1024, 256, 0, stream>>>(bufB,   wp[4], bufA, 64, 128,  8,  8, cKp[4], 1);
  k_conv_lrelu_wmma<3, 4><<<1024, 256, 0, stream>>>(bufA,   wp[5], bufB, 128,128,  8,  8, cKp[5], 0);
  // (1024,128,8,8) == (1024,8192) row-major -> l_fc (1024,256) f32
  k_gemm_bias_wmma<<<dim3(64, 16), 32, 0, stream>>>(bufB, fclwp, fclb, lfc, 1024, 256, 8192);

  // -------- GAT + reductions --------
  k_gat<<<8, 128, 0, stream>>>(lfc, g1W, g1a1, g1a2, g2W, g2a1, g2a2, gat);
  k_finalize<<<8, 544, 0, stream>>>(gfc, lfc, gat, out);
}
